// ConvKANLayer_38405597561235
// MI455X (gfx1250) — matrix operand order repaired and checked by reading, not compiled, not run on hardware
//
#include <hip/hip_runtime.h>
#include <hip/hip_bf16.h>

// ---------------- problem constants ----------------
#define IN_CH   64
#define OUT_CH  128
#define NB      7                 // 1 silu + 6 spline bases per input element
#define FEAT    (IN_CH * NB)      // 448 features per (padded) spatial site
#define KTOT    (9 * FEAT)        // 4032 GEMM reduction dim
#define KSTEPS  (KTOT / 32)       // 126 K=32 steps
#define BATCH   8
#define H       64
#define W       64
#define HP      66                // zero-padded spatial (SAME, 3x3)
#define WP      66
#define NPIX    (BATCH * H * W)   // 32768 GEMM M dim
#define BLK_M   128
#define BLK_N   128
#define LDS_STRIDE 20             // 32 elems (16 dwords) + 4 pad dwords: bank-conflict-free

// Direct global->LDS async staging if the toolchain exposes it (gfx1250)
#if defined(__has_builtin)
#  if __has_builtin(__builtin_amdgcn_global_load_async_to_lds_b128)
#    define USE_ASYNC_LDS 1
#  endif
#endif
#ifndef USE_ASYNC_LDS
#  define USE_ASYNC_LDS 0
#endif

#define AS1 __attribute__((address_space(1)))
#define AS3 __attribute__((address_space(3)))

typedef int v4i __attribute__((ext_vector_type(4)));
typedef AS1 v4i* gv4i_p;          // global int4*
typedef AS3 v4i* lv4i_p;          // LDS int4*

typedef __attribute__((ext_vector_type(16))) __bf16 v16bf;
typedef __attribute__((ext_vector_type(8)))  float  v8f;

union Frag {
    v16bf v;
    uint4 q[2];
};

__device__ __forceinline__ unsigned short f2bf(float f) {
    union { float f; unsigned int u; } cv;
    cv.f = f;
    unsigned int u = cv.u;
    unsigned int r = u + 0x7FFFu + ((u >> 16) & 1u);   // RNE bf16
    return (unsigned short)(r >> 16);
}

__device__ __forceinline__ void wait_async0() {
#if defined(__has_builtin)
#  if __has_builtin(__builtin_amdgcn_s_wait_asynccnt)
    __builtin_amdgcn_s_wait_asynccnt(0);
    return;
#  else
    asm volatile("s_wait_asynccnt 0x0" ::: "memory");
    return;
#  endif
#else
    asm volatile("s_wait_asynccnt 0x0" ::: "memory");
#endif
}

// ---------------------------------------------------------------------------
// Kernel 1: per-element transform -> [silu(v), B0..B5(v)] bf16, zero-padded
// spatial layout xp[n][hp][wp][t*64 + c].  Border sites use v=0 (the
// reference applies b_splines to the SAME-padding zeros too).
// ---------------------------------------------------------------------------
__global__ __launch_bounds__(256)
void kan_transform(const float* __restrict__ x, unsigned short* __restrict__ xp) {
    const int idx = blockIdx.x * blockDim.x + threadIdx.x;
    const int total = BATCH * HP * WP * IN_CH;
    if (idx >= total) return;

    const int c  = idx & (IN_CH - 1);
    const int wp = (idx / IN_CH) % WP;
    const int hp = (idx / (IN_CH * WP)) % HP;
    const int n  = idx / (IN_CH * WP * HP);

    const bool border = (hp == 0) | (hp == HP - 1) | (wp == 0) | (wp == WP - 1);
    float v = 0.0f;
    if (!border)
        v = x[((size_t)((n * H + (hp - 1)) * W + (wp - 1))) * IN_CH + c];

    const float silu = v * (1.0f / (1.0f + __expf(-v)));

    const float h    = 2.0f / 3.0f;
    const float inv1 = 1.0f / h;
    const float inv2 = 1.0f / (2.0f * h);
    const float inv3 = 1.0f / (3.0f * h);

    float b0[9];
#pragma unroll
    for (int i = 0; i < 9; ++i) {
        const float t0 = (float)(i - 3) * h - 1.0f;
        const float t1 = (float)(i - 2) * h - 1.0f;
        b0[i] = ((v >= t0) && (v < t1)) ? 1.0f : 0.0f;
    }
    float b1[8];
#pragma unroll
    for (int i = 0; i < 8; ++i) {
        const float ti  = (float)(i - 3) * h - 1.0f;
        const float ti2 = (float)(i - 1) * h - 1.0f;
        b1[i] = (v - ti) * inv1 * b0[i] + (ti2 - v) * inv1 * b0[i + 1];
    }
    float b2[7];
#pragma unroll
    for (int i = 0; i < 7; ++i) {
        const float ti  = (float)(i - 3) * h - 1.0f;
        const float ti3 = (float)(i)     * h - 1.0f;
        b2[i] = (v - ti) * inv2 * b1[i] + (ti3 - v) * inv2 * b1[i + 1];
    }
    float b3[6];
#pragma unroll
    for (int i = 0; i < 6; ++i) {
        const float ti  = (float)(i - 3) * h - 1.0f;
        const float ti4 = (float)(i + 1) * h - 1.0f;
        b3[i] = (v - ti) * inv3 * b2[i] + (ti4 - v) * inv3 * b2[i + 1];
    }

    unsigned short* dst = xp + ((size_t)((n * HP + hp) * WP + wp)) * FEAT + c;
    dst[0 * IN_CH] = f2bf(silu);
#pragma unroll
    for (int t = 0; t < 6; ++t)
        dst[(t + 1) * IN_CH] = f2bf(b3[t]);
}

// ---------------------------------------------------------------------------
// Kernel 2: weight repack -> bf16 W'[o][k], k = tap*448 + t*64 + c,
// tap = kh*3+kw; reference patch index i = c*9 + tap (channel-major patches).
// ---------------------------------------------------------------------------
__global__ __launch_bounds__(256)
void kan_repack(const float* __restrict__ base_w,
                const float* __restrict__ spline_w,
                unsigned short* __restrict__ wt) {
    const int idx = blockIdx.x * blockDim.x + threadIdx.x;
    if (idx >= OUT_CH * KTOT) return;
    const int k   = idx % KTOT;
    const int o   = idx / KTOT;
    const int tap = k / FEAT;
    const int f   = k % FEAT;
    const int t   = f / IN_CH;
    const int c   = f % IN_CH;
    const int i   = c * 9 + tap;
    const float v = (t == 0) ? base_w[(size_t)o * 576 + i]
                             : spline_w[((size_t)o * 576 + i) * 6 + (t - 1)];
    wt[(size_t)o * KTOT + k] = f2bf(v);
}

// ---------------------------------------------------------------------------
// Kernel 3: LDS-staged double-buffered WMMA GEMM.
// Block = 256 threads (8 waves, 4M x 2N) on a 128x128 tile; wave = 32M x 64N.
// Per K=32 step: stage next A(128x32)/B(128x32) bf16 tiles into LDS (async
// global->LDS when available, else reg-staged), while computing on the
// current buffer: one 12x ds_load_b128 clause feeding 8 WMMAs.
// ---------------------------------------------------------------------------
__global__ __launch_bounds__(256)
void kan_gemm(const unsigned short* __restrict__ xp,
              const unsigned short* __restrict__ wt,
              const float* __restrict__ bias,
              float* __restrict__ out) {
    __shared__ unsigned int ldsA[2][BLK_M][LDS_STRIDE];
    __shared__ unsigned int ldsB[2][BLK_N][LDS_STRIDE];

    const int tid   = threadIdx.x;
    const int lane  = tid & 31;
    const int wave  = tid >> 5;
    const int waveM = wave >> 1;            // 0..3  (32 rows each)
    const int waveN = wave & 1;             // 0..1  (64 cols each)
    const int kq    = lane >> 4;            // K-half selector
    const int l15   = lane & 15;

    // ---- staging addresses: thread t stages 32B of row (t>>1) of A and B ----
    const int sRow = tid >> 1;              // 0..127
    const int sChk = (tid & 1) * 16;        // elem offset of 32B chunk
    const int pm = blockIdx.x * BLK_M + sRow;         // A pixel for staging
    const int pw = pm & (W - 1);
    const int ph = (pm >> 6) & (H - 1);
    const int pn = pm >> 12;
    const unsigned short* aG =
        xp + ((size_t)((pn * HP + ph) * WP + pw)) * FEAT + sChk;
    const unsigned short* bG = wt + (size_t)sRow * KTOT + sChk;
    unsigned int* stA = &ldsA[0][sRow][(tid & 1) * 8];
    unsigned int* stB = &ldsB[0][sRow][(tid & 1) * 8];
    const int bufStrideA = BLK_M * LDS_STRIDE;        // dwords per buffer
    const int bufStrideB = BLK_N * LDS_STRIDE;

    // ---- fragment read addresses ----
    const unsigned int* frA[2];                       // wave's 2 M-subtiles
#pragma unroll
    for (int s = 0; s < 2; ++s)
        frA[s] = &ldsA[0][waveM * 32 + s * 16 + l15][kq * 4];
    const unsigned int* frB[4];                       // wave's 4 N-subtiles
#pragma unroll
    for (int t = 0; t < 4; ++t)
        frB[t] = &ldsB[0][waveN * 64 + t * 16 + l15][kq * 4];

    v8f acc[2][4] = {};

    // ---- prologue: stage k-step 0 into buffer 0 ----
    int ks = 0, tap = 0;
#if USE_ASYNC_LDS
    __builtin_amdgcn_global_load_async_to_lds_b128((gv4i_p)(aG),     (lv4i_p)(stA),     0, 0);
    __builtin_amdgcn_global_load_async_to_lds_b128((gv4i_p)(aG + 8), (lv4i_p)(stA + 4), 0, 0);
    __builtin_amdgcn_global_load_async_to_lds_b128((gv4i_p)(bG),     (lv4i_p)(stB),     0, 0);
    __builtin_amdgcn_global_load_async_to_lds_b128((gv4i_p)(bG + 8), (lv4i_p)(stB + 4), 0, 0);
    aG += 32; bG += 32; ++ks;
    wait_async0();
#else
    {
        uint4 rA0 = *(const uint4*)(aG);
        uint4 rA1 = *(const uint4*)(aG + 8);
        uint4 rB0 = *(const uint4*)(bG);
        uint4 rB1 = *(const uint4*)(bG + 8);
        aG += 32; bG += 32; ++ks;
        *(uint4*)(stA + 0) = rA0;  *(uint4*)(stA + 4) = rA1;
        *(uint4*)(stB + 0) = rB0;  *(uint4*)(stB + 4) = rB1;
    }
#endif
    __syncthreads();

#pragma unroll 1
    for (int iter = 0; iter < KSTEPS; ++iter) {
        const int cur   = iter & 1;
        const bool more = (iter + 1) < KSTEPS;

#if !USE_ASYNC_LDS
        uint4 rA0, rA1, rB0, rB1;
#endif
        // issue next tile's staging (latency hidden behind the WMMAs)
        if (more) {
#if USE_ASYNC_LDS
            const int nxtA = (1 - cur) * bufStrideA;
            const int nxtB = (1 - cur) * bufStrideB;
            __builtin_amdgcn_global_load_async_to_lds_b128((gv4i_p)(aG),     (lv4i_p)(stA + nxtA),     0, 0);
            __builtin_amdgcn_global_load_async_to_lds_b128((gv4i_p)(aG + 8), (lv4i_p)(stA + nxtA + 4), 0, 0);
            __builtin_amdgcn_global_load_async_to_lds_b128((gv4i_p)(bG),     (lv4i_p)(stB + nxtB),     0, 0);
            __builtin_amdgcn_global_load_async_to_lds_b128((gv4i_p)(bG + 8), (lv4i_p)(stB + nxtB + 4), 0, 0);
#else
            rA0 = *(const uint4*)(aG);
            rA1 = *(const uint4*)(aG + 8);
            rB0 = *(const uint4*)(bG);
            rB1 = *(const uint4*)(bG + 8);
#endif
            aG += 32; bG += 32; ++ks;
            if (ks == 14) {                  // tap boundary in A's K walk
                ks = 0; ++tap;
                if (tap == 3 || tap == 6) aG += 63 * FEAT;
            }
        }

        // compute on the current LDS buffer: one 12-load clause, 8 WMMAs
        const int offA = cur * bufStrideA;
        const int offB = cur * bufStrideB;
        Frag a[2], b[4];
#pragma unroll
        for (int t = 0; t < 4; ++t) {
            b[t].q[0] = *(const uint4*)(frB[t] + offB);
            b[t].q[1] = *(const uint4*)(frB[t] + offB + 8);
        }
#pragma unroll
        for (int s = 0; s < 2; ++s) {
            a[s].q[0] = *(const uint4*)(frA[s] + offA);
            a[s].q[1] = *(const uint4*)(frA[s] + offA + 8);
        }
#pragma unroll
        for (int s = 0; s < 2; ++s)
#pragma unroll
            for (int t = 0; t < 4; ++t)
                acc[s][t] = __builtin_amdgcn_wmma_f32_16x16x32_bf16(
                    false, a[s].v, false, b[t].v, (short)0, acc[s][t],
                    false, false);

        // make the next buffer ready before the barrier
        if (more) {
#if USE_ASYNC_LDS
            wait_async0();
#else
            const int nxtA = (1 - cur) * bufStrideA;
            const int nxtB = (1 - cur) * bufStrideB;
            *(uint4*)(stA + nxtA + 0) = rA0;  *(uint4*)(stA + nxtA + 4) = rA1;
            *(uint4*)(stB + nxtB + 0) = rB0;  *(uint4*)(stB + nxtB + 4) = rB1;
#endif
        }
        __syncthreads();
    }

    // ---- epilogue: C/D layout -> VGPR r = (M: r + 8*kq, N: l15) ----
    const int rowAdd = kq * 8;
    const int mWave  = blockIdx.x * BLK_M + waveM * 32;
#pragma unroll
    for (int t = 0; t < 4; ++t) {
        const int col  = waveN * 64 + t * 16 + l15;
        const float bv = bias[col];
#pragma unroll
        for (int s = 0; s < 2; ++s) {
#pragma unroll
            for (int r = 0; r < 8; ++r) {
                out[(size_t)(mWave + s * 16 + rowAdd + r) * OUT_CH + col] =
                    acc[s][t][r] + bv;
            }
        }
    }
}

// ---------------------------------------------------------------------------
extern "C" void kernel_launch(void* const* d_in, const int* in_sizes, int n_in,
                              void* d_out, int out_size, void* d_ws, size_t ws_size,
                              hipStream_t stream) {
    const float* x        = (const float*)d_in[0];
    const float* base_w   = (const float*)d_in[1];
    const float* spline_w = (const float*)d_in[2];
    const float* bias     = (const float*)d_in[3];
    float* out            = (float*)d_out;

    unsigned short* xp = (unsigned short*)d_ws;                   // ~31.2MB bf16
    unsigned short* wt = xp + (size_t)BATCH * HP * WP * FEAT;     // ~1MB bf16

    {
        const int total = BATCH * HP * WP * IN_CH;
        kan_transform<<<(total + 255) / 256, 256, 0, stream>>>(x, xp);
    }
    {
        const int total = OUT_CH * KTOT;
        kan_repack<<<(total + 255) / 256, 256, 0, stream>>>(base_w, spline_w, wt);
    }
    kan_gemm<<<NPIX / BLK_M, 256, 0, stream>>>(xp, wt, bias, out);
}
